// LtiRnnConvConstr_17231408791957
// MI455X (gfx1250) — matrix operand order repaired and checked by reading, not compiled, not run on hardware
//
#include <hip/hip_runtime.h>
#include <hip/hip_bf16.h>

// ---------------- dims ----------------
#define NXD 256
#define NUD 128
#define NYD 128
#define NWD 256
#define BD  256
#define SD  1024
#define K1  384          // NX+NU  (phase 1 K)
#define K2  640          // NX+NU+NW (phase 2 K)
#define N1  256          // NW
#define N2  384          // NY+NX
#define C1N 12           // K1/32
#define C2N 20           // K2/32
#define XROW 648         // 640 + 8 pad (bank-conflict-free rows)

typedef __attribute__((ext_vector_type(16))) __bf16 bf16x16;
typedef __attribute__((ext_vector_type(8)))  float  floatx8;

union FragU { bf16x16 v; uint4 q[2]; };

// ---------------- prep 1: Gauss-Jordan inverse of Y (near-identity) ----------------
__global__ __launch_bounds__(256) void gj_invert(const float* __restrict__ Y,
                                                 float* __restrict__ M) {
  const int tid = threadIdx.x;
  __shared__ float colk[256];
  for (int i = 0; i < 256; ++i) M[i * 256 + tid] = Y[i * 256 + tid];
  __syncthreads();
  for (int k = 0; k < 256; ++k) {
    float pivot = M[k * 256 + k];
    float colv  = M[tid * 256 + k];     // column k, row=tid
    float rowv  = M[k * 256 + tid];     // row k, col=tid
    colk[tid] = colv;
    __syncthreads();
    float rk_new = ((tid == k) ? 1.0f : rowv) / pivot;
    M[k * 256 + tid] = rk_new;          // scaled pivot row (col tid owned by this thread)
    for (int i = 0; i < 256; ++i) {
      if (i == k) continue;
      float f   = colk[i];
      float cur = (tid == k) ? 0.0f : M[i * 256 + tid];
      M[i * 256 + tid] = cur - f * rk_new;
    }
    __syncthreads();
  }
}

// ---------------- prep 2: W1 swizzled [c][n][ki], rows scaled by 1/lambda ----------------
__global__ __launch_bounds__(256) void build_w1(const float* __restrict__ C2,
                                                const float* __restrict__ D21,
                                                const float* __restrict__ lambdas,
                                                const float* __restrict__ b_z,
                                                __bf16* __restrict__ W1s,
                                                float* __restrict__ bz2) {
  int idx = blockIdx.x * 256 + threadIdx.x;           // total C1N*N1*32 = 98304
  if (idx >= C1N * N1 * 32) return;
  int ki = idx & 31;
  int n  = (idx >> 5) & 255;
  int c  = idx >> 13;
  int k  = c * 32 + ki;
  float il = 1.0f / lambdas[n];
  float v  = (k < NXD) ? C2[n * NXD + k] : D21[n * NUD + (k - NXD)];
  W1s[idx] = (__bf16)(v * il);
  if (c == 0 && ki == 0) bz2[n] = b_z[n] * il;
}

// ---------------- prep 3: W2 swizzled; x-rows folded with Yinv ----------------
__global__ __launch_bounds__(256) void build_w2(const float* __restrict__ C1,
                                                const float* __restrict__ D11,
                                                const float* __restrict__ D12,
                                                const float* __restrict__ A_w,
                                                const float* __restrict__ B1_w,
                                                const float* __restrict__ B2_w,
                                                const float* __restrict__ b_y,
                                                const float* __restrict__ b_x,
                                                const float* __restrict__ Yinv,
                                                __bf16* __restrict__ W2s,
                                                float* __restrict__ bias2) {
  int idx = blockIdx.x * 256 + threadIdx.x;           // total C2N*N2*32 = 245760
  if (idx >= C2N * N2 * 32) return;
  int ki = idx & 31;
  int n  = (idx >> 5) % N2;
  int c  = idx / (N2 * 32);
  int k  = c * 32 + ki;
  float v;
  if (n < NYD) {                                      // y output rows
    v = (k < NXD)  ? C1[n * NXD + k]
      : (k < 384)  ? D11[n * NUD + (k - NXD)]
                   : D12[n * NWD + (k - 384)];
  } else {                                            // x' rows: (Yinv^T @ [A|B1|B2])
    int r = n - NYD;
    float s = 0.0f;
    for (int j = 0; j < NXD; ++j) {
      float mjk = (k < NXD) ? A_w[j * NXD + k]
                : (k < 384) ? B1_w[j * NUD + (k - NXD)]
                            : B2_w[j * NWD + (k - 384)];
      s += Yinv[j * 256 + r] * mjk;
    }
    v = s;
  }
  W2s[idx] = (__bf16)v;
  if (c == 0 && ki == 0) {
    if (n < NYD) bias2[n] = b_y[n];
    else {
      int r = n - NYD;
      float s = 0.0f;
      for (int j = 0; j < NXD; ++j) s += b_x[j] * Yinv[j * 256 + r];
      bias2[n] = s;
    }
  }
}

// ---------------- fragment loaders ----------------
__device__ __forceinline__ bf16x16 load_afrag(const __bf16* arow, int kb) {
  FragU a;
  a.q[0] = *(const uint4*)(arow + kb);        // K = kb+h16 + 0..7
  a.q[1] = *(const uint4*)(arow + kb + 8);    // K = kb+h16 + 8..15
  return a.v;
}
__device__ __forceinline__ bf16x16 load_bfrag(const __bf16* p) {
  FragU b;
  b.q[0] = ((const uint4*)p)[0];
  b.q[1] = ((const uint4*)p)[1];
  return b.v;
}
#define WMMA_BF16(A, Bm, C) \
  __builtin_amdgcn_wmma_f32_16x16x32_bf16(false, (A), false, (Bm), (short)0, (C), false, false)

// ---------------- main persistent scan kernel: 16 blocks x 256 threads ----------------
__global__ __launch_bounds__(256) void lti_main(const float* __restrict__ xpred,
                                                const __bf16* __restrict__ W1s,
                                                const __bf16* __restrict__ W2s,
                                                const float* __restrict__ bz2,
                                                const float* __restrict__ bias2,
                                                float* __restrict__ yout,
                                                float* __restrict__ xfin,
                                                float* __restrict__ xfin2) {
  __shared__ __bf16 X[16 * XROW];               // [16 rows][ x(256) | u(128) | w(256) | pad ]
  const int tid  = threadIdx.x;
  const int lane = tid & 31;
  const int wave = tid >> 5;
  const int b0   = blockIdx.x * 16;
  const int mrow = lane & 15;                   // A-frag row / B-frag column-in-tile
  const int hv   = lane >> 4;                   // half-wave select
  const int mbase = hv * 8;                     // C/D fragment row base

  // zero x state
  for (int e = 0; e < 16; ++e) {
    int idx = tid * 16 + e;
    X[(idx >> 8) * XROW + (idx & 255)] = (__bf16)0.0f;
  }

  const __bf16* arow = X + mrow * XROW + hv * 16;

  // per-wave column assignments
  const int nc1 = wave * 32;                    // phase 1: 2 x 16 cols
  const int nc2 = wave * 48;                    // phase 2: 3 x 16 cols
  const int n1a = nc1 + mrow, n1b = n1a + 16;
  const float bzA = bz2[n1a], bzB = bz2[n1b];
  int   n2c[3]; float bb2[3]; bool isx[3];
#pragma unroll
  for (int t = 0; t < 3; ++t) {
    n2c[t] = nc2 + 16 * t + mrow;
    bb2[t] = bias2[n2c[t]];
    isx[t] = (nc2 + 16 * t) >= NYD;
  }
  const __bf16* p1 = W1s + (size_t)((nc1 + mrow) * 32 + hv * 16);
  const __bf16* p2 = W2s + (size_t)((nc2 + mrow) * 32 + hv * 16);

  // u staging: thread t handles row (t>>4), 8 consecutive floats
  const int um  = tid >> 4;
  const int uk0 = (tid & 15) * 8;
  const float* ubase = xpred + (size_t)(b0 + um) * SD * NUD + uk0;

  // preload u(0) into registers
  float ur[8];
#pragma unroll
  for (int q = 0; q < 8; ++q) ur[q] = ubase[q];

  for (int s = 0; s < SD; ++s) {
    // commit staged u(s) -> LDS cols [256,384) as bf16 (cheap: regs -> ds)
#pragma unroll
    for (int q = 0; q < 8; ++q) X[um * XROW + NXD + uk0 + q] = (__bf16)ur[q];
    __syncthreads();                            // u + previous x' visible

    // ---- phase 1: z = [x,u] @ W1^T ----  (A double-buffered)
    floatx8 a0 = {}, a1 = {};
    bf16x16 af = load_afrag(arow, 0);
#pragma unroll
    for (int c = 0; c < C1N; ++c) {
      bf16x16 afn = af;
      if (c + 1 < C1N) afn = load_afrag(arow, (c + 1) * 32);
      a0 = WMMA_BF16(af, load_bfrag(p1 + c * (N1 * 32)), a0);
      a1 = WMMA_BF16(af, load_bfrag(p1 + c * (N1 * 32) + 512), a1);
      af = afn;
    }
    // epilogue: w = tanh(z + bz') -> LDS cols [384,640)
#pragma unroll
    for (int r = 0; r < 8; ++r) {
      int m = mbase + r;
      X[m * XROW + K1 + n1a] = (__bf16)tanhf(a0[r] + bzA);
      X[m * XROW + K1 + n1b] = (__bf16)tanhf(a1[r] + bzB);
    }
    __syncthreads();                            // w visible

    // prefetch u(s+1) into registers; latency hidden behind phase-2 WMMAs
    float urn[8];
    if (s + 1 < SD) {
      const float* up = ubase + (size_t)(s + 1) * NUD;
#pragma unroll
      for (int q = 0; q < 8; ++q) urn[q] = up[q];
    } else {
#pragma unroll
      for (int q = 0; q < 8; ++q) urn[q] = 0.0f;
    }

    // ---- phase 2: [y | x'] = [x,u,w] @ W2^T ----  (A + B double-buffered)
    floatx8 d0 = {}, d1 = {}, d2 = {};
    bf16x16 af2 = load_afrag(arow, 0);
    bf16x16 bf0 = load_bfrag(p2);
    bf16x16 bf1 = load_bfrag(p2 + 512);
    bf16x16 bf2v = load_bfrag(p2 + 1024);
#pragma unroll 2
    for (int c = 0; c < C2N; ++c) {
      bf16x16 afn = af2, bn0 = bf0, bn1 = bf1, bn2 = bf2v;
      if (c + 1 < C2N) {
        const __bf16* pc = p2 + (size_t)(c + 1) * (N2 * 32);
        afn = load_afrag(arow, (c + 1) * 32);
        bn0 = load_bfrag(pc);
        bn1 = load_bfrag(pc + 512);
        bn2 = load_bfrag(pc + 1024);
      }
      d0 = WMMA_BF16(af2, bf0,  d0);
      d1 = WMMA_BF16(af2, bf1,  d1);
      d2 = WMMA_BF16(af2, bf2v, d2);
      af2 = afn; bf0 = bn0; bf1 = bn1; bf2v = bn2;
    }

    // epilogue: y -> global, x' -> registers
    float xv[3][8];
#pragma unroll
    for (int t = 0; t < 3; ++t) {
      floatx8 acc = (t == 0) ? d0 : (t == 1) ? d1 : d2;
      if (!isx[t]) {
#pragma unroll
        for (int r = 0; r < 8; ++r)
          yout[((size_t)(b0 + mbase + r) * SD + s) * NYD + n2c[t]] = acc[r] + bb2[t];
      } else {
#pragma unroll
        for (int r = 0; r < 8; ++r) xv[t][r] = acc[r] + bb2[t];
      }
    }
    __syncthreads();                            // all reads of old x done
#pragma unroll
    for (int t = 0; t < 3; ++t) {
      if (isx[t]) {
        int n = n2c[t] - NYD;
#pragma unroll
        for (int r = 0; r < 8; ++r)
          X[(mbase + r) * XROW + n] = (__bf16)xv[t][r];
      }
    }
#pragma unroll
    for (int q = 0; q < 8; ++q) ur[q] = urn[q];
    // top-of-loop barrier makes x' visible for next step
  }

  __syncthreads();
  for (int e = 0; e < 16; ++e) {
    int idx = tid * 16 + e;
    int m = idx >> 8, n = idx & 255;
    float v = (float)X[m * XROW + n];
    xfin [(size_t)(b0 + m) * NXD + n] = v;
    xfin2[(size_t)(b0 + m) * NXD + n] = v;
  }
}

// ---------------- launcher ----------------
extern "C" void kernel_launch(void* const* d_in, const int* in_sizes, int n_in,
                              void* d_out, int out_size, void* d_ws, size_t ws_size,
                              hipStream_t stream) {
  const float* xpred   = (const float*)d_in[0];
  const float* Y       = (const float*)d_in[1];
  const float* A_w     = (const float*)d_in[2];
  const float* B1_w    = (const float*)d_in[3];
  const float* B2_w    = (const float*)d_in[4];
  const float* C1_w    = (const float*)d_in[5];
  const float* D11_w   = (const float*)d_in[6];
  const float* D12_w   = (const float*)d_in[7];
  const float* C2_w    = (const float*)d_in[8];
  const float* D21_w   = (const float*)d_in[9];
  const float* lambdas = (const float*)d_in[10];
  const float* b_z     = (const float*)d_in[11];
  const float* b_y     = (const float*)d_in[12];
  const float* b_x     = (const float*)d_in[13];

  char* ws = (char*)d_ws;
  float*  Yinv  = (float*)(ws + 0);            // 262144 B
  float*  bz2   = (float*)(ws + 262144);       // 1024 B
  float*  bias2 = (float*)(ws + 263168);       // 1536 B
  __bf16* W1s   = (__bf16*)(ws + 264704);      // 196608 B
  __bf16* W2s   = (__bf16*)(ws + 461312);      // 491520 B (end ~0.95 MB)

  gj_invert<<<1, 256, 0, stream>>>(Y, Yinv);
  build_w1<<<(C1N * N1 * 32) / 256, 256, 0, stream>>>(C2_w, D21_w, lambdas, b_z, W1s, bz2);
  build_w2<<<(C2N * N2 * 32) / 256, 256, 0, stream>>>(C1_w, D11_w, D12_w, A_w, B1_w, B2_w,
                                                      b_y, b_x, Yinv, W2s, bias2);

  float* yout  = (float*)d_out;
  float* xfin  = yout + (size_t)BD * SD * NYD;
  float* xfin2 = xfin + (size_t)BD * NXD;
  lti_main<<<16, 256, 0, stream>>>(xpred, W1s, W2s, bz2, bias2, yout, xfin, xfin2);
}
